// TPGGMLLinear_41317585388031
// MI455X (gfx1250) — matrix operand order repaired and checked by reading, not compiled, not run on hardware
//
#include <hip/hip_runtime.h>

typedef __attribute__((ext_vector_type(16))) _Float16 v16h;
typedef __attribute__((ext_vector_type(8)))  _Float16 v8h;
typedef __attribute__((ext_vector_type(4)))  _Float16 v4h;
typedef __attribute__((ext_vector_type(8)))  float    v8f;
typedef __attribute__((ext_vector_type(4)))  float    v4f;
typedef __attribute__((ext_vector_type(4)))  int      v4i;

namespace {
constexpr int BM   = 128;  // output rows per workgroup
constexpr int BN   = 128;  // output cols per workgroup
constexpr int BK   = 64;   // K per stage (= 2 Q8 blocks = 2 WMMA K-steps)
constexpr int LDSP = 72;   // padded LDS row stride (halves) to spread banks
}

// GEMM: out[M,N] = x[M,K] * W[N,K]^T + bias, W dequantized Q8_0-style on the fly.
// M=B*S, N=OUT, K=IN. Assumes M%BM==0, N%BN==0, K%BK==0 (holds for 8192/4096/4096).
//
// 256 threads = 8 waves (wave32). Waves tile 4(M) x 2(N); each wave computes a
// 32x64 patch = 2x4 WMMA 16x16 f32 accumulators -> 16 v_wmma per K-stage.
// Software pipeline: global loads for stage s+1 are issued into registers
// before the WMMA block of stage s, hiding HBM latency behind matrix math.
__global__ __launch_bounds__(256)
void q8_linear_wmma_kernel(const float* __restrict__ x,
                           const float* __restrict__ q_scales,  // [N, n_blocks]
                           const int*   __restrict__ q_vals,    // [N, n_blocks*32]
                           const float* __restrict__ bias,      // [N]
                           float* __restrict__ out,             // [M, N]
                           int M, int N, int K, int n_blocks)
{
  __shared__ _Float16 As[BM][LDSP];  // x tile, f16, [m][k]     (18 KB)
  __shared__ _Float16 Bs[BN][LDSP];  // dequant W tile, f16, [n][k] (18 KB)

  const int tid   = threadIdx.x;
  const int lane  = tid & 31;        // wave32
  const int wave  = tid >> 5;        // 0..7
  const int waveM = wave & 3;        // 0..3 -> 32-row band
  const int waveN = wave >> 2;       // 0..1 -> 64-col band

  const int n0 = blockIdx.x * BN;
  const int m0 = blockIdx.y * BM;

  // Stage-load mapping: 2 threads per row, 32 elems each (8x 16-byte loads).
  // A: 128 rows x 64 f32; B: 128 rows x 64 int32 codes (one Q8 block/thread).
  const int ldRow  = tid >> 1;         // 0..127
  const int ldKoff = (tid & 1) * 32;   // 0 or 32

  const float* gA = x      + (size_t)(m0 + ldRow) * K + ldKoff;
  const int*   gQ = q_vals + (size_t)(n0 + ldRow) * K + ldKoff;   // K == n_blocks*32
  const float* gS = q_scales + (size_t)(n0 + ldRow) * n_blocks + (ldKoff >> 5);

  // ---- prologue: stage-0 global loads into registers ----
  v4f aReg[8];
  v4i bReg[8];
  float dReg = gS[0];
#pragma unroll
  for (int i = 0; i < 8; ++i) aReg[i] = *(const v4f*)(gA + 4 * i);
#pragma unroll
  for (int i = 0; i < 8; ++i) bReg[i] = *(const v4i*)(gQ + 4 * i);

  v8f acc[2][4];
#pragma unroll
  for (int mi = 0; mi < 2; ++mi)
#pragma unroll
    for (int ni = 0; ni < 4; ++ni)
      acc[mi][ni] = (v8f)0.0f;

  const int laneN = lane & 15;
  const int aKl   = (lane & 16) ? 8 : 0;   // A frag K-half select (ISA layout)
  const int bKl   = (lane & 16) ? 8 : 0;   // B frag K-half select (ISA layout)

  const int nStages = K / BK;
  for (int s = 0; s < nStages; ++s) {
    // ---- staged registers -> LDS (f32->f16 convert, Q8 dequant) ----
#pragma unroll
    for (int i = 0; i < 8; ++i) {
      v4h h;
      h[0] = (_Float16)aReg[i][0]; h[1] = (_Float16)aReg[i][1];
      h[2] = (_Float16)aReg[i][2]; h[3] = (_Float16)aReg[i][3];
      *(v4h*)&As[ldRow][ldKoff + 4 * i] = h;
    }
    {
      const float d = dReg;
#pragma unroll
      for (int i = 0; i < 8; ++i) {
        v4h h;
        h[0] = (_Float16)((float)(bReg[i][0] - 128) * d);
        h[1] = (_Float16)((float)(bReg[i][1] - 128) * d);
        h[2] = (_Float16)((float)(bReg[i][2] - 128) * d);
        h[3] = (_Float16)((float)(bReg[i][3] - 128) * d);
        *(v4h*)&Bs[ldRow][ldKoff + 4 * i] = h;
      }
    }

    // ---- issue next-stage global loads (latency hidden behind WMMAs) ----
    if (s + 1 < nStages) {
      const float* gA2 = gA + (size_t)(s + 1) * BK;
      const int*   gQ2 = gQ + (size_t)(s + 1) * BK;
#pragma unroll
      for (int i = 0; i < 8; ++i) aReg[i] = *(const v4f*)(gA2 + 4 * i);
#pragma unroll
      for (int i = 0; i < 8; ++i) bReg[i] = *(const v4i*)(gQ2 + 4 * i);
      dReg = gS[2 * (s + 1)];   // BK/32 = 2 Q8 blocks per stage
    }

    __syncthreads();

    // ---- compute: 2 WMMA K=32 steps, 2x4 tiles -> 16 v_wmma per stage ----
#pragma unroll
    for (int kk = 0; kk < BK; kk += 32) {
      v16h a[2], b[4];
#pragma unroll
      for (int mi = 0; mi < 2; ++mi) {
        // A 16x32 f16 layout: lane m = lane&15; halves 0..7 = K[koff..koff+8),
        // halves 8..15 = K[16+koff..16+koff+8), koff = (lane>=16)?8:0.
        const _Float16* p = &As[waveM * 32 + mi * 16 + laneN][kk + aKl];
        v8h lo = *(const v8h*)p;
        v8h hi = *(const v8h*)(p + 16);
        a[mi] = __builtin_shufflevector(lo, hi, 0,1,2,3,4,5,6,7,8,9,10,11,12,13,14,15);
      }
#pragma unroll
      for (int ni = 0; ni < 4; ++ni) {
        // B 32x16 f16 layout: lane n = lane&15; halves 0..7 = K[koff..koff+8),
        // halves 8..15 = K[16+koff..16+koff+8), koff = (lane>=16)?8:0.
        const _Float16* p = &Bs[waveN * 64 + ni * 16 + laneN][kk + bKl];
        v8h lo = *(const v8h*)p;
        v8h hi = *(const v8h*)(p + 16);
        b[ni] = __builtin_shufflevector(lo, hi, 0,1,2,3,4,5,6,7,8,9,10,11,12,13,14,15);
      }
#pragma unroll
      for (int mi = 0; mi < 2; ++mi)
#pragma unroll
        for (int ni = 0; ni < 4; ++ni)
          acc[mi][ni] = __builtin_amdgcn_wmma_f32_16x16x32_f16(
              false, a[mi], false, b[ni], (short)0, acc[mi][ni], false, false);
    }

    __syncthreads();
  }

  // ---- epilogue: C/D layout -> global, add bias ----
  // C f32 16x16: lane n = lane&15; VGPR r holds m = r + ((lane>=16)?8:0).
  const int mHalf = (lane & 16) ? 8 : 0;
#pragma unroll
  for (int ni = 0; ni < 4; ++ni) {
    const int   n  = n0 + waveN * 64 + ni * 16 + laneN;
    const float bv = bias[n];
#pragma unroll
    for (int mi = 0; mi < 2; ++mi) {
#pragma unroll
      for (int r = 0; r < 8; ++r) {
        const int m = m0 + waveM * 32 + mi * 16 + mHalf + r;
        out[(size_t)m * N + n] = acc[mi][ni][r] + bv;
      }
    }
  }
}

extern "C" void kernel_launch(void* const* d_in, const int* in_sizes, int n_in,
                              void* d_out, int out_size, void* d_ws, size_t ws_size,
                              hipStream_t stream) {
  (void)d_ws; (void)ws_size; (void)n_in; (void)out_size;
  const float* x        = (const float*)d_in[0];
  const float* q_scales = (const float*)d_in[1];
  const int*   q_vals   = (const int*)d_in[2];
  const float* bias     = (const float*)d_in[3];
  float*       out      = (float*)d_out;

  const int OUT      = in_sizes[3];            // bias length
  const int n_blocks = in_sizes[1] / OUT;      // q_scales = [OUT, n_blocks]
  const int K        = n_blocks * 32;          // IN
  const int M        = in_sizes[0] / K;        // B*S

  dim3 block(256, 1, 1);
  dim3 grid((unsigned)(OUT / BN), (unsigned)(M / BM), 1);
  q8_linear_wmma_kernel<<<grid, block, 0, stream>>>(
      x, q_scales, q_vals, bias, out, M, OUT, K, n_blocks);
}
